// StructureBottleneck_74345883894074
// MI455X (gfx1250) — matrix operand order repaired and checked by reading, not compiled, not run on hardware
//
#include <hip/hip_runtime.h>
#include <cstdint>
#include <cstddef>

// ---------------------------------------------------------------- types
typedef _Float16 f16;
typedef __attribute__((ext_vector_type(16))) _Float16 v16h;
typedef __attribute__((ext_vector_type(8)))  _Float16 v8h;
typedef __attribute__((ext_vector_type(8)))  float    v8f;
typedef __attribute__((ext_vector_type(4)))  unsigned int v4u;
typedef __attribute__((ext_vector_type(8)))  int      v8i;
typedef __attribute__((ext_vector_type(4)))  int      v4i;

#if defined(__has_builtin)
# if __has_builtin(__builtin_amdgcn_tensor_load_to_lds) && __has_builtin(__builtin_amdgcn_s_wait_tensorcnt)
#  define USE_TDM 1
# endif
#endif
#ifndef USE_TDM
# define USE_TDM 0
#endif

// ---------------------------------------------------------------- sizes
constexpr int B_   = 4;
constexpr int L_   = 2048;
constexpr int D_   = 512;
constexpr int M_   = B_ * L_;        // 8192 token rows
constexpr int PLU  = 496;
constexpr int CH1  = 248;
constexpr int CURV = 16;
constexpr int CIN_W = 3 * D_ + CURV; // 1552
constexpr int SIN_W = 21;            // 10 LN-stats + 3 ss_prob + 8 ss_lp
constexpr int BOT  = 64;
constexpr int RS   = 16;             // rowstats slots per row
constexpr float TAU   = 1.5f;        // TAU_PAIR == TAU_ATTN
constexpr float TAU_SSc = 1.25f;
constexpr float NEGV  = -10000.f;

// ---------------------------------------------------------------- helpers
__device__ inline float gelu_f(float x) {
    return 0.5f * x * (1.f + erff(x * 0.70710678118654752f));
}

__device__ inline void top3_ins(float v, int id,
                                float& v0, int& i0, float& v1, int& i1,
                                float& v2, int& i2) {
    bool b0 = (v > v0) || (v == v0 && id < i0);
    bool b1 = (v > v1) || (v == v1 && id < i1);
    bool b2 = (v > v2) || (v == v2 && id < i2);
    if (b0)      { v2 = v1; i2 = i1; v1 = v0; i1 = i0; v0 = v; i0 = id; }
    else if (b1) { v2 = v1; i2 = i1; v1 = v;  i1 = id; }
    else if (b2) { v2 = v;  i2 = id; }
}

// A-operand K index for v_wmma_f32_16x16x32_f16 (16x32 f16 A layout)
__device__ inline int a_koff(int t, int hi) {
    int r = t >> 1, pos = t & 1;
    return ((r < 4) ? 2 * r : 16 + 2 * (r - 4)) + pos + hi * 8;
}

// ---------------------------------------------------------------- staging kernels
// H f32 -> content_in cols [0,512) f16
__global__ __launch_bounds__(256)
void cvt_h_kernel(const float* __restrict__ Hg, f16* __restrict__ cin, int ldcin) {
    int idx = blockIdx.x * 256 + threadIdx.x;
    if (idx >= M_ * D_) return;
    int row = idx / D_, d = idx - row * D_;
    cin[(size_t)row * ldcin + d] = (f16)Hg[idx];
}

__global__ __launch_bounds__(256)
void cvt_kappa_kernel(const float* __restrict__ src, f16* __restrict__ dst, int n) {
    int idx = blockIdx.x * 256 + threadIdx.x;
    if (idx < n) dst[idx] = (f16)src[idx];
}

__global__ __launch_bounds__(256)
void mask_f_kernel(const uint8_t* __restrict__ mask, float* __restrict__ mf, int n) {
    int idx = blockIdx.x * 256 + threadIdx.x;
    if (idx < n) mf[idx] = mask[idx] ? 1.f : 0.f;
}

// LDS-tiled transpose: Hg (B,L,D) f32 -> H16T (B,D,L) f16, both sides coalesced
__global__ __launch_bounds__(256)
void transpose_h_kernel(const float* __restrict__ Hg, f16* __restrict__ H16T) {
    __shared__ float tile[32][33];
    int b  = blockIdx.z;
    int i0 = blockIdx.y * 32;   // L index
    int d0 = blockIdx.x * 32;   // D index
    int tid = threadIdx.x;
    for (int e = tid; e < 1024; e += 256) {
        int r = e >> 5, c = e & 31;
        tile[r][c] = Hg[((size_t)b * L_ + i0 + r) * D_ + d0 + c];
    }
    __syncthreads();
    for (int e = tid; e < 1024; e += 256) {
        int r = e >> 5, c = e & 31;
        H16T[((size_t)b * D_ + d0 + r) * L_ + i0 + c] = (f16)tile[c][r];
    }
}

// ---------------------------------------------------------------- pass 1: fused row reductions
// one wave per (b,i) row of pair_logits, float4 loads, branch-free selects.
__global__ __launch_bounds__(32)
void row_stats_kernel(const float* __restrict__ pair,
                      const float* __restrict__ mf,
                      float* __restrict__ rst) {
    int i = blockIdx.x;
    int b = blockIdx.y;
    int lane = threadIdx.x;
    const float* prow = pair + ((size_t)b * L_ + i) * L_;
    const float* mfb  = mf + (size_t)b * L_;
    bool mi = mfb[i] != 0.f;

    float mrun = -3.4e38f, s = 0.f, t = 0.f;
    float maxx = -3.4e38f, sp = 0.f, lpm = 0.f, dpm = 0.f;
    float nv = 0.f, nl = 0.f, nd = 0.f;
    float tv0 = -3.4e38f, tv1 = -3.4e38f, tv2 = -3.4e38f;
    int   ti0 = 0x7fffffff, ti1 = 0x7fffffff, ti2 = 0x7fffffff;

    for (int c = lane; c < L_ / 4; c += 32) {
        int jb = c * 4;
        __builtin_prefetch(prow + jb + 512, 0, 3);
        float4 x4 = *(const float4*)(prow + jb);
        float4 m4 = *(const float4*)(mfb + jb);
        float xs[4] = {x4.x, x4.y, x4.z, x4.w};
        float ms[4] = {m4.x, m4.y, m4.z, m4.w};
#pragma unroll
        for (int u = 0; u < 4; ++u) {
            int j = jb + u;
            bool valid = mi && (ms[u] != 0.f) && (j != i);
            float x = valid ? xs[u] : NEGV;
            float y = x * (1.f / TAU);
            if (y > mrun) { float sc = __expf(mrun - y); s *= sc; t *= sc; mrun = y; }
            float e = __expf(y - mrun);
            s += e; t += e * y;
            maxx = fmaxf(maxx, x);
            float p = 1.f / (1.f + __expf(-y));   // sigmoid(x / TAU_PAIR)
            sp += p;
            int dist = (j > i) ? (j - i) : (i - j);
            if (dist <= 8) { lpm += p; nl += valid ? 1.f : 0.f; }
            else           { dpm += p; nd += valid ? 1.f : 0.f; }
            nv += valid ? 1.f : 0.f;
            top3_ins(x, j, tv0, ti0, tv1, ti1, tv2, ti2);
        }
    }
    // wave32 butterfly reduction
    for (int o = 16; o > 0; o >>= 1) {
        float om = __shfl_xor(mrun, o);
        float os = __shfl_xor(s, o);
        float ot = __shfl_xor(t, o);
        float nm = fmaxf(mrun, om);
        float sa = __expf(mrun - nm), sb = __expf(om - nm);
        s = s * sa + os * sb;
        t = t * sa + ot * sb;
        mrun = nm;
        maxx = fmaxf(maxx, __shfl_xor(maxx, o));
        sp  += __shfl_xor(sp, o);
        lpm += __shfl_xor(lpm, o);
        dpm += __shfl_xor(dpm, o);
        nv  += __shfl_xor(nv, o);
        nl  += __shfl_xor(nl, o);
        nd  += __shfl_xor(nd, o);
        float ov0 = __shfl_xor(tv0, o); int oi0 = __shfl_xor(ti0, o);
        float ov1 = __shfl_xor(tv1, o); int oi1 = __shfl_xor(ti1, o);
        float ov2 = __shfl_xor(tv2, o); int oi2 = __shfl_xor(ti2, o);
        top3_ins(ov0, oi0, tv0, ti0, tv1, ti1, tv2, ti2);
        top3_ins(ov1, oi1, tv0, ti0, tv1, ti1, tv2, ti2);
        top3_ins(ov2, oi2, tv0, ti0, tv1, ti1, tv2, ti2);
    }
    if (lane == 0) {
        float* r = rst + ((size_t)b * L_ + i) * RS;
        float lse = mrun + __logf(s);
        r[0] = lse;               // log-sum-exp of clean/tau
        r[1] = lse - t / s;       // exact softmax entropy
        r[2] = maxx;              // row max of clean
        r[3] = sp;                // sum sigmoid
        r[4] = lpm; r[5] = dpm;   // local / distal sigmoid mass
        r[6] = nv;  r[7] = nl; r[8] = nd;
        r[9] = tv0; r[10] = tv1; r[11] = tv2;
        r[12] = __int_as_float(ti0);
        r[13] = __int_as_float(ti1);
        r[14] = __int_as_float(ti2);
        r[15] = 0.f;
    }
}

// ---------------------------------------------------------------- pass 2: ctx_mean = softmax(clean) @ H  (flash-style, WMMA)
// H staged TRANSPOSED in LDS (tileT[d*32 + j]) via TDM so B fragments are
// two contiguous 16B LDS loads per subtile.
__global__ __launch_bounds__(256)
void attn_ctx_kernel(const float* __restrict__ pair,
                     const f16* __restrict__ H16T,
                     const float* __restrict__ mf,
                     const float* __restrict__ rst,
                     f16* __restrict__ ctx, int ldctx) {
    __shared__ f16 tileT[D_ * 32];      // 32 KB: 512 d-rows x 32 j-cols
    int tid  = threadIdx.x;
    int lane = tid & 31, wave = tid >> 5;
    int hi = lane >> 4, lo = lane & 15;
    int b  = blockIdx.y;
    int i0 = blockIdx.x * 16;
    int d0 = wave * 64;                 // each wave owns 64 D-columns
    int irow = i0 + lo;
    const float* mfb = mf + (size_t)b * L_;
    bool mi = mfb[irow] != 0.f;
    float lse = rst[((size_t)b * L_ + irow) * RS + 0];
    const float* prow = pair + ((size_t)b * L_ + irow) * L_;

    v8f acc[4] = {};
    for (int j0 = 0; j0 < L_; j0 += 32) {
        // ---- stage transposed tile: rows d=0..511, cols j0..j0+31
#if USE_TDM
        if (tid < 32) {   // wave 0 issues one TDM descriptor for the tile
            unsigned long long ga =
                (unsigned long long)(const void*)(H16T + (size_t)b * D_ * L_ + j0);
            unsigned int la = (unsigned int)(unsigned long long)(const void*)tileT;
            v4u g0;
            g0[0] = 1u;                                          // count=1
            g0[1] = la;                                          // lds_addr
            g0[2] = (unsigned int)(ga & 0xffffffffull);          // global lo
            g0[3] = (unsigned int)((ga >> 32) & 0x01ffffffull) | (2u << 30); // hi + type=2
            v8i g1;
            g1[0] = (1 << 16);                                   // data_size = 2B
            g1[1] = (int)((L_ & 0xffff) << 16);                  // tensor_dim0 lo
            g1[2] = (int)(((unsigned)L_ >> 16) | ((unsigned)(D_ & 0xffff) << 16)); // dim0 hi|dim1 lo
            g1[3] = (int)((32u & 0xffff) << 16);                 // dim1 hi(0) | tile_dim0=32
            g1[4] = D_;                                          // tile_dim1=512, tile_dim2=0
            g1[5] = L_;                                          // dim0 stride lo
            g1[6] = 0;
            g1[7] = 0;
            v4i z4 = {0, 0, 0, 0};
            v8i z8 = {0, 0, 0, 0, 0, 0, 0, 0};
            __builtin_amdgcn_tensor_load_to_lds(g0, g1, z4, z4, z8, 0);
            __builtin_amdgcn_s_wait_tensorcnt(0);
        }
        __syncthreads();
        asm volatile("" ::: "memory");
#else
        for (int r = tid; r < D_; r += 256) {   // 512 rows x 64B
            const uint4* src = (const uint4*)(H16T + ((size_t)(b * D_ + r)) * L_ + j0);
            uint4* dst = (uint4*)(tileT + r * 32);
            dst[0] = src[0]; dst[1] = src[1]; dst[2] = src[2]; dst[3] = src[3];
        }
        __syncthreads();
#endif
        __builtin_prefetch(prow + j0 + 32, 0, 3);

        // ---- A operand: p = exp(clean/tau - lse); vector loads + selects.
        // lane layout: a[0..7] = k hi*8+0..7 ; a[8..15] = k 16+hi*8+0..7
        float xs[16], ms[16];
        {
            float4 xa = *(const float4*)(prow + j0 + hi * 8);
            float4 xb = *(const float4*)(prow + j0 + hi * 8 + 4);
            float4 xc = *(const float4*)(prow + j0 + 16 + hi * 8);
            float4 xd = *(const float4*)(prow + j0 + 16 + hi * 8 + 4);
            float4 ma = *(const float4*)(mfb + j0 + hi * 8);
            float4 mb = *(const float4*)(mfb + j0 + hi * 8 + 4);
            float4 mc = *(const float4*)(mfb + j0 + 16 + hi * 8);
            float4 md = *(const float4*)(mfb + j0 + 16 + hi * 8 + 4);
            xs[0]=xa.x; xs[1]=xa.y; xs[2]=xa.z; xs[3]=xa.w;
            xs[4]=xb.x; xs[5]=xb.y; xs[6]=xb.z; xs[7]=xb.w;
            xs[8]=xc.x; xs[9]=xc.y; xs[10]=xc.z; xs[11]=xc.w;
            xs[12]=xd.x; xs[13]=xd.y; xs[14]=xd.z; xs[15]=xd.w;
            ms[0]=ma.x; ms[1]=ma.y; ms[2]=ma.z; ms[3]=ma.w;
            ms[4]=mb.x; ms[5]=mb.y; ms[6]=mb.z; ms[7]=mb.w;
            ms[8]=mc.x; ms[9]=mc.y; ms[10]=mc.z; ms[11]=mc.w;
            ms[12]=md.x; ms[13]=md.y; ms[14]=md.z; ms[15]=md.w;
        }
        v16h a;
#pragma unroll
        for (int t = 0; t < 16; ++t) {
            int j = j0 + t + ((t < 8) ? 0 : 8) + hi * 8;
            bool valid = mi && (ms[t] != 0.f) && (j != irow);
            float x = valid ? xs[t] : NEGV;
            a[t] = (f16)__expf(x * (1.f / TAU) - lse);
        }
        // ---- 4 WMMAs; B fragment = 2 contiguous v8h LDS loads per subtile
#pragma unroll
        for (int s4 = 0; s4 < 4; ++s4) {
            int n = d0 + s4 * 16 + lo;
            v8h b0 = *(const v8h*)(tileT + n * 32 + hi * 8);
            v8h b1 = *(const v8h*)(tileT + n * 32 + hi * 8 + 16);
            v16h bm;
#pragma unroll
            for (int t = 0; t < 8; ++t) { bm[t] = b0[t]; bm[8 + t] = b1[t]; }
            acc[s4] = __builtin_amdgcn_wmma_f32_16x16x32_f16(
                false, a, false, bm, (short)0, acc[s4], false, false);
        }
        __syncthreads();
    }
    // ---- epilogue: C layout m = r + hi*8, n = lane&15
#pragma unroll
    for (int s4 = 0; s4 < 4; ++s4)
#pragma unroll
        for (int r = 0; r < 8; ++r) {
            int grow = b * L_ + i0 + r + hi * 8;
            int dcol = d0 + s4 * 16 + lo;
            ctx[(size_t)grow * ldctx + dcol] = (f16)acc[s4][r];
        }
}

// ---------------------------------------------------------------- top-3 gather context
__global__ __launch_bounds__(256)
void topk_ctx_kernel(const float* __restrict__ Hg,
                     const float* __restrict__ rst,
                     f16* __restrict__ ctx, int ldctx) {
    int row = blockIdx.x;              // 0..M_-1
    int b = row / L_;
    const float* r = rst + (size_t)row * RS;
    float v0 = r[9], v1 = r[10], v2 = r[11];
    int i0 = __float_as_int(r[12]); i0 = min(max(i0, 0), L_ - 1);
    int i1 = __float_as_int(r[13]); i1 = min(max(i1, 0), L_ - 1);
    int i2 = __float_as_int(r[14]); i2 = min(max(i2, 0), L_ - 1);
    float g0 = (v0 > -9000.f) ? 1.f : 0.f;
    float g1 = (v1 > -9000.f) ? 1.f : 0.f;
    float g2 = (v2 > -9000.f) ? 1.f : 0.f;
    float l0 = (g0 != 0.f ? v0 : NEGV) * (1.f / TAU);
    float l1 = (g1 != 0.f ? v1 : NEGV) * (1.f / TAU);
    float l2 = (g2 != 0.f ? v2 : NEGV) * (1.f / TAU);
    float mx = fmaxf(l0, fmaxf(l1, l2));
    float w0 = __expf(l0 - mx) * g0;
    float w1 = __expf(l1 - mx) * g1;
    float w2 = __expf(l2 - mx) * g2;
    float sw = fmaxf(w0 + w1 + w2, 1e-8f);
    w0 /= sw; w1 /= sw; w2 /= sw;
    const float* h0 = Hg + ((size_t)b * L_ + i0) * D_;
    const float* h1 = Hg + ((size_t)b * L_ + i1) * D_;
    const float* h2 = Hg + ((size_t)b * L_ + i2) * D_;
    for (int d = threadIdx.x; d < D_; d += 256) {
        float c = w0 * h0[d] + w1 * h1[d] + w2 * h2[d];
        ctx[(size_t)row * ldctx + d] = (f16)c;
    }
}

// ---------------------------------------------------------------- per-token scalar stats (part 1)
__global__ __launch_bounds__(256)
void ssfeat_kernel(const float* __restrict__ ss_logits,
                   const float* __restrict__ sslp_w, const float* __restrict__ sslp_b,
                   const float* __restrict__ rst,
                   float* __restrict__ ssf, float* __restrict__ up,
                   float* __restrict__ pen, float* __restrict__ stats_out) {
    int row = blockIdx.x * 256 + threadIdx.x;
    if (row >= M_) return;
    const float* sl = ss_logits + (size_t)row * 3;
    float y0 = sl[0] / TAU_SSc, y1 = sl[1] / TAU_SSc, y2 = sl[2] / TAU_SSc;
    float m = fmaxf(y0, fmaxf(y1, y2));
    float e0 = __expf(y0 - m), e1 = __expf(y1 - m), e2 = __expf(y2 - m);
    float z = e0 + e1 + e2;
    float p0 = e0 / z, p1 = e1 / z, p2 = e2 / z;
    float ssent = -(p0 * __logf(p0 + 1e-8f) + p1 * __logf(p1 + 1e-8f) +
                    p2 * __logf(p2 + 1e-8f));
    float* sf = ssf + (size_t)row * 11;
    sf[0] = p0; sf[1] = p1; sf[2] = p2;
#pragma unroll
    for (int o = 0; o < 8; ++o)
        sf[3 + o] = sl[0] * sslp_w[o * 3 + 0] + sl[1] * sslp_w[o * 3 + 1] +
                    sl[2] * sslp_w[o * 3 + 2] + sslp_b[o];
    const float* r = rst + (size_t)row * RS;
    float nv = fmaxf(r[6], 1.f);
    float pm = r[3] / nv;
    float mpp = 1.f / (1.f + __expf(-r[2] * (1.f / TAU)));
    float unp = 1.f - mpp;
    float logn = fmaxf(__logf(nv), 1e-4f);
    float pe = r[1] / logn;
    if (!(pe == pe)) pe = 0.f;
    pe = fminf(fmaxf(pe, 0.f), 1.f);
    float lm = r[4] / fmaxf(r[7], 1.f);
    float dm = r[5] / fmaxf(r[8], 1.f);
    up[row] = unp; pen[row] = pe;
    float* so = stats_out + (size_t)row * 10;
    so[0] = pm; so[1] = mpp; so[2] = unp; so[3] = pe; so[4] = ssent;
    so[5] = lm; so[6] = dm;
}

// ---------------------------------------------------------------- windowed means + mask + LN (part 2)
__global__ __launch_bounds__(256)
void stats2_kernel(const float* __restrict__ mf,
                   const float* __restrict__ up, const float* __restrict__ pen,
                   const float* __restrict__ ssf,
                   const float* __restrict__ ln_g, const float* __restrict__ ln_b,
                   float* __restrict__ stats_out, f16* __restrict__ sin_) {
    int row = blockIdx.x * 256 + threadIdx.x;
    if (row >= M_) return;
    int b = row / L_, i = row - b * L_;
    const float* mfb = mf + (size_t)b * L_;
    auto wmean = [&](const float* v, int rad) {
        float sx = 0.f, sm = 0.f;
        for (int dj = -rad; dj <= rad; ++dj) {
            int j = i + dj;
            if (j < 0 || j >= L_) continue;
            float m = mfb[j];
            sx += v[(size_t)b * L_ + j] * m;
            sm += m;
        }
        return sx / fmaxf(sm, 1.f);
    };
    float s[10];
    float* so = stats_out + (size_t)row * 10;
#pragma unroll
    for (int c = 0; c < 7; ++c) s[c] = so[c];
    s[7] = wmean(up, 2);
    s[8] = wmean(up, 5);
    s[9] = wmean(pen, 5);
    float mfi = mfb[i];
#pragma unroll
    for (int c = 0; c < 10; ++c) {
        float v = s[c] * mfi;
        if (!(v == v)) v = 0.f;
        else if (isinf(v)) v = v > 0.f ? 1.f : 0.f;
        s[c] = v;
        so[c] = v;                       // final stats output
    }
    float mean = 0.f;
#pragma unroll
    for (int c = 0; c < 10; ++c) mean += s[c];
    mean *= 0.1f;
    float var = 0.f;
#pragma unroll
    for (int c = 0; c < 10; ++c) { float d = s[c] - mean; var += d * d; }
    var *= 0.1f;
    float inv = rsqrtf(var + 1e-5f);
    f16* si = sin_ + (size_t)row * SIN_W;
#pragma unroll
    for (int c = 0; c < 10; ++c)
        si[c] = (f16)((s[c] - mean) * inv * ln_g[c] + ln_b[c]);
    const float* sf = ssf + (size_t)row * 11;
#pragma unroll
    for (int k = 0; k < 11; ++k) si[10 + k] = (f16)sf[k];
}

// ---------------------------------------------------------------- generic WMMA GEMM: C = act(A @ W^T + b)
// A: f16 MxK (lda), W: f32 NxK (torch Linear layout), out f16/f32 with ldc.
// Fast path uses vector loads when the k-window is fully in-bounds & aligned.
template <int ACT, bool OUT16>
__global__ __launch_bounds__(128)
void wmma_gemm(const f16* __restrict__ A, int lda,
               const float* __restrict__ W, const float* __restrict__ bias,
               void* __restrict__ outp, int ldc,
               int M, int N, int K,
               const uint8_t* __restrict__ rowmask) {
    int lane = threadIdx.x & 31;
    int wave = threadIdx.x >> 5;
    int hi = lane >> 4, lo = lane & 15;
    int m0 = blockIdx.x * 16;
    int n0 = blockIdx.y * 64 + wave * 16;
    if (n0 >= N) return;
    int mr = m0 + lo;
    int nc = n0 + lo;
    bool aligned = ((lda & 7) == 0) && ((K & 3) == 0);
    v8f acc = {};
    for (int k0 = 0; k0 < K; k0 += 32) {
        v16h a, bm;
        if (aligned && (k0 + 32 <= K)) {
            // A: two contiguous 16B half-vectors per lane
            const f16* ap = A + (size_t)mr * lda + k0 + hi * 16;
            v8h a0 = *(const v8h*)ap;
            v8h a1 = *(const v8h*)(ap + 8);
#pragma unroll
            for (int t = 0; t < 8; ++t) { a[t] = a0[t]; a[8 + t] = a1[t]; }
            if (nc < N) {
                const float* wp = W + (size_t)nc * K + k0 + hi * 16;
                float4 w0 = *(const float4*)(wp);
                float4 w1 = *(const float4*)(wp + 4);
                float4 w2 = *(const float4*)(wp + 8);
                float4 w3 = *(const float4*)(wp + 12);
                bm[0]=(f16)w0.x; bm[1]=(f16)w0.y; bm[2]=(f16)w0.z; bm[3]=(f16)w0.w;
                bm[4]=(f16)w1.x; bm[5]=(f16)w1.y; bm[6]=(f16)w1.z; bm[7]=(f16)w1.w;
                bm[8]=(f16)w2.x; bm[9]=(f16)w2.y; bm[10]=(f16)w2.z; bm[11]=(f16)w2.w;
                bm[12]=(f16)w3.x; bm[13]=(f16)w3.y; bm[14]=(f16)w3.z; bm[15]=(f16)w3.w;
            } else {
#pragma unroll
                for (int t = 0; t < 16; ++t) bm[t] = (f16)0.f;
            }
        } else {
#pragma unroll
            for (int t = 0; t < 16; ++t) {
                int ka = k0 + hi * 16 + t;
                a[t] = (ka < K) ? A[(size_t)mr * lda + ka] : (f16)0.f;
                int kb = k0 + hi * 16 + t;
                bm[t] = (kb < K && nc < N) ? (f16)W[(size_t)nc * K + kb] : (f16)0.f;
            }
        }
        acc = __builtin_amdgcn_wmma_f32_16x16x32_f16(
            false, a, false, bm, (short)0, acc, false, false);
    }
#pragma unroll
    for (int r = 0; r < 8; ++r) {
        int m = m0 + r + hi * 8;
        int n = n0 + lo;
        if (m < M && n < N) {
            float x = acc[r] + bias[n];
            if (ACT == 1) x = gelu_f(x);
            if (rowmask) x *= (float)rowmask[m];
            if (OUT16) ((f16*)outp)[(size_t)m * ldc + n] = (f16)x;
            else       ((float*)outp)[(size_t)m * ldc + n] = x;
        }
    }
}

// ---------------------------------------------------------------- host orchestration
extern "C" void kernel_launch(void* const* d_in, const int* in_sizes, int n_in,
                              void* d_out, int out_size, void* d_ws, size_t ws_size,
                              hipStream_t stream) {
    (void)in_sizes; (void)n_in; (void)out_size; (void)ws_size;
    const float*   Hg      = (const float*)d_in[0];
    const float*   pair    = (const float*)d_in[1];
    const float*   ssl     = (const float*)d_in[2];
    const float*   kap     = (const float*)d_in[3];
    const uint8_t* mask    = (const uint8_t*)d_in[4];
    const float* curv_w1 = (const float*)d_in[5];  const float* curv_b1 = (const float*)d_in[6];
    const float* curv_w2 = (const float*)d_in[7];  const float* curv_b2 = (const float*)d_in[8];
    const float* sslp_w  = (const float*)d_in[9];  const float* sslp_b  = (const float*)d_in[10];
    const float* ln_g    = (const float*)d_in[11]; const float* ln_b    = (const float*)d_in[12];
    const float* cont_w1 = (const float*)d_in[13]; const float* cont_b1 = (const float*)d_in[14];
    const float* cont_w2 = (const float*)d_in[15]; const float* cont_b2 = (const float*)d_in[16];
    const float* stat_w1 = (const float*)d_in[17]; const float* stat_b1 = (const float*)d_in[18];
    const float* stat_w2 = (const float*)d_in[19]; const float* stat_b2 = (const float*)d_in[20];
    const float* mrg_w1  = (const float*)d_in[21]; const float* mrg_b1  = (const float*)d_in[22];
    const float* mrg_w2  = (const float*)d_in[23]; const float* mrg_b2  = (const float*)d_in[24];

    float* out       = (float*)d_out;
    float* btok      = out;                        // M_ x 64
    float* stats_out = out + (size_t)M_ * BOT;     // M_ x 10

    char* p = (char*)d_ws;
    auto alloc = [&](size_t bytes) {
        char* q = p; p += (bytes + 255) & ~(size_t)255; return q;
    };
    f16*   H16T  = (f16*)  alloc((size_t)M_ * D_ * 2);      // (B,D,L) transposed
    f16*   CINb  = (f16*)  alloc((size_t)M_ * CIN_W * 2);   // content_in staging
    float* RST   = (float*)alloc((size_t)M_ * RS * 4);
    float* MF    = (float*)alloc((size_t)M_ * 4);           // mask as f32
    float* UP    = (float*)alloc((size_t)M_ * 4);
    float* PEN   = (float*)alloc((size_t)M_ * 4);
    float* SSF   = (float*)alloc((size_t)M_ * 11 * 4);
    f16*   SIN   = (f16*)  alloc((size_t)M_ * SIN_W * 2);
    f16*   KAP16 = (f16*)  alloc((size_t)M_ * PLU * 2);
    f16*   CT1   = (f16*)  alloc((size_t)M_ * CH1 * 2);
    f16*   CHID  = (f16*)  alloc((size_t)M_ * 128 * 2);
    f16*   MIN_  = (f16*)  alloc((size_t)M_ * 96 * 2);      // merge_in staging
    f16*   SHID  = (f16*)  alloc((size_t)M_ * 64 * 2);
    f16*   MHID  = (f16*)  alloc((size_t)M_ * 128 * 2);

    // staging conversions
    mask_f_kernel<<<(M_ + 255) / 256, 256, 0, stream>>>(mask, MF, M_);
    cvt_h_kernel<<<(M_ * D_ + 255) / 256, 256, 0, stream>>>(Hg, CINb, CIN_W);
    transpose_h_kernel<<<dim3(D_ / 32, L_ / 32, B_), 256, 0, stream>>>(Hg, H16T);
    cvt_kappa_kernel<<<(M_ * PLU + 255) / 256, 256, 0, stream>>>(kap, KAP16, M_ * PLU);

    // pass 1: fused row reductions over pair_logits
    row_stats_kernel<<<dim3(L_, B_), 32, 0, stream>>>(pair, MF, RST);

    // pass 2: softmax @ H via WMMA, writes content_in cols [512,1024)
    attn_ctx_kernel<<<dim3(L_ / 16, B_), 256, 0, stream>>>(
        pair, H16T, MF, RST, CINb + D_, CIN_W);

    // top-3 context, cols [1024,1536)
    topk_ctx_kernel<<<M_, 256, 0, stream>>>(Hg, RST, CINb + 2 * D_, CIN_W);

    // curvature MLP -> cols [1536,1552)
    wmma_gemm<1, true><<<dim3(M_ / 16, (CH1 + 63) / 64), 128, 0, stream>>>(
        KAP16, PLU, curv_w1, curv_b1, CT1, CH1, M_, CH1, PLU, nullptr);
    wmma_gemm<0, true><<<dim3(M_ / 16, 1), 128, 0, stream>>>(
        CT1, CH1, curv_w2, curv_b2, CINb + 3 * D_, CIN_W, M_, CURV, CH1, nullptr);

    // scalar stats + windowed means + layernorm
    ssfeat_kernel<<<(M_ + 255) / 256, 256, 0, stream>>>(
        ssl, sslp_w, sslp_b, RST, SSF, UP, PEN, stats_out);
    stats2_kernel<<<(M_ + 255) / 256, 256, 0, stream>>>(
        MF, UP, PEN, SSF, ln_g, ln_b, stats_out, SIN);

    // content MLP
    wmma_gemm<1, true><<<dim3(M_ / 16, 2), 128, 0, stream>>>(
        CINb, CIN_W, cont_w1, cont_b1, CHID, 128, M_, 128, CIN_W, nullptr);
    wmma_gemm<0, true><<<dim3(M_ / 16, 1), 128, 0, stream>>>(
        CHID, 128, cont_w2, cont_b2, MIN_, 96, M_, 64, 128, nullptr);

    // stats MLP
    wmma_gemm<1, true><<<dim3(M_ / 16, 1), 128, 0, stream>>>(
        SIN, SIN_W, stat_w1, stat_b1, SHID, 64, M_, 64, SIN_W, nullptr);
    wmma_gemm<0, true><<<dim3(M_ / 16, 1), 128, 0, stream>>>(
        SHID, 64, stat_w2, stat_b2, MIN_ + 64, 96, M_, 32, 64, nullptr);

    // merge MLP -> B_tok (masked)
    wmma_gemm<1, true><<<dim3(M_ / 16, 2), 128, 0, stream>>>(
        MIN_, 96, mrg_w1, mrg_b1, MHID, 128, M_, 128, 96, nullptr);
    wmma_gemm<0, false><<<dim3(M_ / 16, 1), 128, 0, stream>>>(
        MHID, 128, mrg_w2, mrg_b2, btok, BOT, M_, BOT, 128, mask);
}
